// SimpleGRU_17437567222244
// MI455X (gfx1250) — matrix operand order repaired and checked by reading, not compiled, not run on hardware
//
#include <hip/hip_runtime.h>
#include <stdint.h>

// Problem dims (match reference)
#define B_    64
#define T_    1024
#define DIN_  512
#define H_    1024
#define O_    512
#define NBLK_SCAN 48   // 48 blocks * 8 waves = 384 waves == 384 scan GEMM tasks

typedef __attribute__((ext_vector_type(16))) __bf16 v16bf;
typedef __attribute__((ext_vector_type(8)))  float  v8f;
typedef unsigned int u32x4 __attribute__((ext_vector_type(4)));
typedef unsigned int uint32x4 __attribute__((ext_vector_type(4)));
typedef int int32x4 __attribute__((ext_vector_type(4)));
typedef int int32x8 __attribute__((ext_vector_type(8)));

union Frag { u32x4 u[2]; v16bf v; };   // 32 bytes = 16 bf16 = one WMMA A/B operand

__device__ __forceinline__ unsigned short f2bf(float f) {
  unsigned int u = __float_as_uint(f);
  u += 0x7FFFu + ((u >> 16) & 1u);            // round-to-nearest-even
  return (unsigned short)(u >> 16);
}
__device__ __forceinline__ float bf2f(unsigned short s) {
  return __uint_as_float(((unsigned int)s) << 16);
}
__device__ __forceinline__ float sigmoidf_(float x) {
  return 1.0f / (1.0f + __expf(-x));
}

// ---------------- elementwise prep kernels ----------------

__global__ void cvt_f32_bf16(const float* __restrict__ s,
                             unsigned short* __restrict__ d, int n) {
  int i = blockIdx.x * blockDim.x + threadIdx.x;
  if (i < n) d[i] = f2bf(s[i]);
}

// src: [K,N] row-major f32  ->  dst: [N,K] row-major bf16 (i.e. B^T for NT GEMM)
__global__ void transpose_f32_bf16(const float* __restrict__ s,
                                   unsigned short* __restrict__ d, int K, int N) {
  int i = blockIdx.x * blockDim.x + threadIdx.x;
  if (i < K * N) {
    int n = i / K, k = i - n * K;
    d[i] = f2bf(s[(size_t)k * N + n]);
  }
}

__global__ void init_state(float* __restrict__ h, unsigned short* __restrict__ hb,
                           int* __restrict__ bar, int n) {
  int i = blockIdx.x * blockDim.x + threadIdx.x;
  if (i < n) { h[i] = 0.0f; hb[i] = 0; }
  if (i < 2) bar[i] = 0;
}

// ---------------- generic bf16 WMMA GEMM:  C[M,N] = A[M,K] * Bt[N,K]^T + bias -------------
// One wave per 16x64 tile (4 f32 accumulators); the A fragment is loaded once per K-chunk
// and reused by 4 WMMAs -> 4x compute per A-load. All fragment loads are contiguous b128.

template <int OUT_BF16>
__global__ __launch_bounds__(256) void gemm_bf16_nt(
    const unsigned short* __restrict__ A,    // [M,K] bf16 row-major
    const unsigned short* __restrict__ Bt,   // [N,K] bf16 row-major (transposed B)
    const float* __restrict__ bias,          // [N]
    void* __restrict__ Cout,                 // [M,N] f32 or bf16
    int M, int N, int K) {
  const int wave = threadIdx.x >> 5;
  const int lane = threadIdx.x & 31;
  const int lrow = lane & 15;
  const int lhi  = lane >> 4;
  const long tile  = (long)blockIdx.x * 8 + wave;
  const int  ntn4  = N >> 6;                  // 64-wide N tiles
  const long total = (long)(M >> 4) * ntn4;
  if (tile >= total) return;                  // uniform per-wave -> EXEC stays all-ones
  const int mt  = (int)(tile / ntn4);
  const int nt4 = (int)(tile % ntn4);
  const int m  = mt * 16 + lrow;
  const int n0 = nt4 * 64;
  const unsigned short* arow = A + (size_t)m * K;
  const unsigned short* brow0 = Bt + (size_t)(n0 +  0 + lrow) * K;
  const unsigned short* brow1 = Bt + (size_t)(n0 + 16 + lrow) * K;
  const unsigned short* brow2 = Bt + (size_t)(n0 + 32 + lrow) * K;
  const unsigned short* brow3 = Bt + (size_t)(n0 + 48 + lrow) * K;
  v8f acc0 = {}, acc1 = {}, acc2 = {}, acc3 = {};
  for (int kc = 0; kc < K; kc += 32) {
    Frag fa, b0, b1, b2, b3;
    // A 16x32 bf16, contiguous k order used here: lanes 0-15 K 0-15, lanes 16-31 K 16-31
    fa.u[0] = *reinterpret_cast<const u32x4*>(arow + kc + lhi * 16);
    fa.u[1] = *reinterpret_cast<const u32x4*>(arow + kc + lhi * 16 + 8);
    // B 32x16 bf16, same contiguous k order as A: lanes 0-15 K 0-15, lanes 16-31 K 16-31
    const int kb = kc + lhi * 16;
    b0.u[0] = *reinterpret_cast<const u32x4*>(brow0 + kb);
    b0.u[1] = *reinterpret_cast<const u32x4*>(brow0 + kb + 8);
    b1.u[0] = *reinterpret_cast<const u32x4*>(brow1 + kb);
    b1.u[1] = *reinterpret_cast<const u32x4*>(brow1 + kb + 8);
    b2.u[0] = *reinterpret_cast<const u32x4*>(brow2 + kb);
    b2.u[1] = *reinterpret_cast<const u32x4*>(brow2 + kb + 8);
    b3.u[0] = *reinterpret_cast<const u32x4*>(brow3 + kb);
    b3.u[1] = *reinterpret_cast<const u32x4*>(brow3 + kb + 8);
    acc0 = __builtin_amdgcn_wmma_f32_16x16x32_bf16(false, fa.v, false, b0.v, (short)0, acc0, false, false);
    acc1 = __builtin_amdgcn_wmma_f32_16x16x32_bf16(false, fa.v, false, b1.v, (short)0, acc1, false, false);
    acc2 = __builtin_amdgcn_wmma_f32_16x16x32_bf16(false, fa.v, false, b2.v, (short)0, acc2, false, false);
    acc3 = __builtin_amdgcn_wmma_f32_16x16x32_bf16(false, fa.v, false, b3.v, (short)0, acc3, false, false);
  }
  // C/D layout: reg r -> row r + 8*(lane>=16), col = lane%16
  const int rbase = mt * 16 + lhi * 8;
  v8f accs[4] = {acc0, acc1, acc2, acc3};
#pragma unroll
  for (int j = 0; j < 4; ++j) {
    const int n = n0 + j * 16 + lrow;
    const float bv = bias ? bias[n] : 0.0f;
    if (OUT_BF16) {
      unsigned short* C = (unsigned short*)Cout;
#pragma unroll
      for (int r = 0; r < 8; ++r)
        C[(size_t)(rbase + r) * N + n] = f2bf(accs[j][r] + bv);
    } else {
      float* C = (float*)Cout;
#pragma unroll
      for (int r = 0; r < 8; ++r)
        C[(size_t)(rbase + r) * N + n] = accs[j][r] + bv;
    }
  }
}

// ---------------- persistent GRU scan kernel ----------------

__device__ __forceinline__ void grid_sync(int* bar) {
  __syncthreads();
  if (threadIdx.x == 0) {
    __threadfence();
    int gen  = __hip_atomic_load(bar + 1, __ATOMIC_ACQUIRE, __HIP_MEMORY_SCOPE_AGENT);
    int prev = __hip_atomic_fetch_add(bar, 1, __ATOMIC_ACQ_REL, __HIP_MEMORY_SCOPE_AGENT);
    if (prev == NBLK_SCAN - 1) {
      __hip_atomic_store(bar, 0, __ATOMIC_RELAXED, __HIP_MEMORY_SCOPE_AGENT);
      __hip_atomic_fetch_add(bar + 1, 1, __ATOMIC_RELEASE, __HIP_MEMORY_SCOPE_AGENT);
    } else {
      while (__hip_atomic_load(bar + 1, __ATOMIC_ACQUIRE, __HIP_MEMORY_SCOPE_AGENT) == gen) {
        __builtin_amdgcn_s_sleep(1);
      }
    }
  }
  __syncthreads();
}

// 48 blocks x 256 threads (8 waves) persistent. Per step t:
//   TDM tensor_load_to_lds stages h (128 KB bf16, flat 64x1024x2B 2D tile) into LDS,
//   issued once by wave 0, tracked by TENSORcnt -> A fragments come from ds_load_b128
//   tmp[g] = h @ W_hg : 384 16x32 WMMA tasks == 384 waves (one task per wave, no stragglers)
//   pointwise GRU update -> h (f32), h (bf16 mirror), hs[:,t,:] (bf16)
__global__ __launch_bounds__(256) void gru_scan(
    const unsigned short* __restrict__ xr,    // [B,T,H] bf16 precomputed gates
    const unsigned short* __restrict__ xz,
    const unsigned short* __restrict__ xn,
    const unsigned short* __restrict__ whrT,  // [H,H] bf16, transposed
    const unsigned short* __restrict__ whzT,
    const unsigned short* __restrict__ whnT,
    const float* __restrict__ b_hn,           // [H]
    float* __restrict__ h,                    // [B,H] f32 carry
    unsigned short* __restrict__ hb,          // [B,H] bf16 mirror
    unsigned short* __restrict__ hs,          // [B,T,H] bf16 history
    float* __restrict__ tmp,                  // [3,B,H] f32 matmul results
    int* __restrict__ bar) {
  extern __shared__ unsigned short hlds[];    // B_*H_ bf16 = 128 KB (of 320 KB WGP LDS)
  const int wave = threadIdx.x >> 5;
  const int lane = threadIdx.x & 31;
  const int lrow = lane & 15;
  const int lhi  = lane >> 4;
  const int gw   = blockIdx.x * 8 + wave;     // global wave id, 0..383

  // ---- TDM descriptor (D#) for hb[64x1024 bf16] -> LDS, built per ISA 8.3/8.4 ----
  const unsigned long long ga = (unsigned long long)hb;
  const unsigned lds_off = (unsigned)(unsigned long long)(uintptr_t)hlds; // LDS byte offset
  uint32x4 g0;
  g0[0] = 1u;                                   // count=1 valid descriptor
  g0[1] = lds_off;                              // lds_addr[31:0]
  g0[2] = (unsigned)ga;                         // global_addr[31:0]
  g0[3] = ((unsigned)(ga >> 32) & 0x01FFFFFFu)  // global_addr[56:32]
          | 0x80000000u;                        // type=2 ("image")
  int32x8 g1;
  g1[0] = 0x00010000;                           // data_size=1 (2 bytes/elem)
  g1[1] = (int)(1024u << 16);                   // tensor_dim0[15:0]=1024 in bits 63:48
  g1[2] = (int)(64u << 16);                     // tensor_dim1[15:0]=64 in bits 95:80
  g1[3] = (int)(1024u << 16);                   // tile_dim0=1024 in bits 127:112
  g1[4] = 64;                                   // tile_dim1=64 (tile_dim2=0)
  g1[5] = 1024;                                 // tensor_dim0_stride[31:0]=1024
  g1[6] = 0;                                    // stride0 hi / stride1 lo
  g1[7] = 0;
  int32x4 g2 = {0, 0, 0, 0};                    // 2D tensor: groups 2/3 unused
  int32x4 g3 = {0, 0, 0, 0};

  // task decode (wave-uniform, fixed for whole scan): 3 gates * 4 mtiles * 32 n2tiles
  const int gate = gw >> 7;                   // 0..2
  const int rem  = gw & 127;
  const int mt   = rem >> 5;                  // 0..3  (16 B-rows each)
  const int nt2  = rem & 31;                  // 0..31 (32 H-cols each)
  const unsigned short* W = (gate == 0) ? whrT : (gate == 1) ? whzT : whnT;
  const int m  = mt * 16 + lrow;
  const int n0 = nt2 * 32;
  const unsigned short* brow0 = W + (size_t)(n0 + lrow) * H_;
  const unsigned short* brow1 = W + (size_t)(n0 + 16 + lrow) * H_;

  for (int t = 0; t < T_; ++t) {
    // --- stage h into LDS via Tensor Data Mover (async DMA, TENSORcnt-tracked) ---
    if (wave == 0) {
#if defined(__clang_major__) && __clang_major__ >= 23
      int32x8 g4 = {0, 0, 0, 0, 0, 0, 0, 0};
      __builtin_amdgcn_tensor_load_to_lds(g0, g1, g2, g3, g4, 0);
#else
      __builtin_amdgcn_tensor_load_to_lds(g0, g1, g2, g3, 0);
#endif
      __builtin_amdgcn_s_wait_tensorcnt(0);
    }
    __syncthreads();

    // --- recurrent GEMM: one 16x32 tile per wave, A from LDS, B from L2-resident weights ---
    {
      const unsigned short* arow = hlds + (size_t)m * H_;
      v8f acc0 = {}, acc1 = {};
#pragma unroll 4
      for (int kc = 0; kc < H_; kc += 32) {
        Frag fa, b0, b1;
        fa.u[0] = *reinterpret_cast<const u32x4*>(arow + kc + lhi * 16);       // ds_load_b128
        fa.u[1] = *reinterpret_cast<const u32x4*>(arow + kc + lhi * 16 + 8);
        const int kb = kc + lhi * 16;
        b0.u[0] = *reinterpret_cast<const u32x4*>(brow0 + kb);
        b0.u[1] = *reinterpret_cast<const u32x4*>(brow0 + kb + 8);
        b1.u[0] = *reinterpret_cast<const u32x4*>(brow1 + kb);
        b1.u[1] = *reinterpret_cast<const u32x4*>(brow1 + kb + 8);
        acc0 = __builtin_amdgcn_wmma_f32_16x16x32_bf16(false, fa.v, false, b0.v, (short)0, acc0, false, false);
        acc1 = __builtin_amdgcn_wmma_f32_16x16x32_bf16(false, fa.v, false, b1.v, (short)0, acc1, false, false);
      }
      float* out = tmp + (size_t)gate * (B_ * H_) + (size_t)(mt * 16 + lhi * 8) * H_;
#pragma unroll
      for (int r = 0; r < 8; ++r) out[(size_t)r * H_ + n0 + lrow] = acc0[r];
#pragma unroll
      for (int r = 0; r < 8; ++r) out[(size_t)r * H_ + n0 + 16 + lrow] = acc1[r];
    }
    grid_sync(bar);

    // --- pointwise GRU cell update (f32 math) ---
    for (int e = blockIdx.x * 256 + threadIdx.x; e < B_ * H_; e += NBLK_SCAN * 256) {
      const int b   = e >> 10;                // H_ == 1024
      const int col = e & (H_ - 1);
      const size_t gidx = ((size_t)b * T_ + t) * H_ + col;
      const float r  = sigmoidf_(bf2f(xr[gidx]) + tmp[e]);
      const float z  = sigmoidf_(bf2f(xz[gidx]) + tmp[B_ * H_ + e]);
      const float nn = tanhf(bf2f(xn[gidx]) + r * (tmp[2 * B_ * H_ + e] + b_hn[col]));
      const float hnew = (1.0f - z) * nn + z * h[e];
      h[e] = hnew;
      const unsigned short hv = f2bf(hnew);
      hb[e] = hv;
      hs[gidx] = hv;
    }
    grid_sync(bar);
  }
}

// ---------------- host launch ----------------

extern "C" void kernel_launch(void* const* d_in, const int* in_sizes, int n_in,
                              void* d_out, int out_size, void* d_ws, size_t ws_size,
                              hipStream_t stream) {
  (void)in_sizes; (void)n_in; (void)out_size; (void)ws_size;
  const float* x     = (const float*)d_in[0];
  const float* w_ir  = (const float*)d_in[1];
  const float* w_iz  = (const float*)d_in[2];
  const float* w_in  = (const float*)d_in[3];
  const float* b_ir  = (const float*)d_in[4];
  const float* b_iz  = (const float*)d_in[5];
  const float* b_in  = (const float*)d_in[6];
  const float* w_hr  = (const float*)d_in[7];
  const float* w_hz  = (const float*)d_in[8];
  const float* w_hn  = (const float*)d_in[9];
  const float* b_hn  = (const float*)d_in[10];
  const float* w_out = (const float*)d_in[11];
  const float* b_out = (const float*)d_in[12];
  float* out = (float*)d_out;

  const size_t BT = (size_t)B_ * T_;
  char* ws = (char*)d_ws;
  size_t off = 0;
  auto wsalloc = [&](size_t bytes) -> char* {
    char* p = ws + off;
    off += (bytes + 255) & ~(size_t)255;
    return p;
  };
  unsigned short* xbf  = (unsigned short*)wsalloc(BT * DIN_ * 2);        // x in bf16
  unsigned short* wirT = (unsigned short*)wsalloc((size_t)H_ * DIN_ * 2);
  unsigned short* wizT = (unsigned short*)wsalloc((size_t)H_ * DIN_ * 2);
  unsigned short* winT = (unsigned short*)wsalloc((size_t)H_ * DIN_ * 2);
  unsigned short* whrT = (unsigned short*)wsalloc((size_t)H_ * H_ * 2);
  unsigned short* whzT = (unsigned short*)wsalloc((size_t)H_ * H_ * 2);
  unsigned short* whnT = (unsigned short*)wsalloc((size_t)H_ * H_ * 2);
  unsigned short* woT  = (unsigned short*)wsalloc((size_t)O_ * H_ * 2);
  unsigned short* gxr  = (unsigned short*)wsalloc(BT * H_ * 2);          // precomputed gates
  unsigned short* gxz  = (unsigned short*)wsalloc(BT * H_ * 2);
  unsigned short* gxn  = (unsigned short*)wsalloc(BT * H_ * 2);
  unsigned short* hsb  = (unsigned short*)wsalloc(BT * H_ * 2);          // hidden history bf16
  float*          hf   = (float*)wsalloc((size_t)B_ * H_ * 4);
  unsigned short* hbm  = (unsigned short*)wsalloc((size_t)B_ * H_ * 2);
  float*          tmp  = (float*)wsalloc((size_t)3 * B_ * H_ * 4);
  int*            bar  = (int*)wsalloc(256);

  // 1. convert x to bf16
  {
    int n = (int)(BT * DIN_);
    cvt_f32_bf16<<<(n + 255) / 256, 256, 0, stream>>>(x, xbf, n);
  }
  // 2. transpose + convert all weights to bf16 [N,K]
  {
    int n1 = DIN_ * H_;  // input weights [512,1024] -> [1024,512]
    transpose_f32_bf16<<<(n1 + 255) / 256, 256, 0, stream>>>(w_ir, wirT, DIN_, H_);
    transpose_f32_bf16<<<(n1 + 255) / 256, 256, 0, stream>>>(w_iz, wizT, DIN_, H_);
    transpose_f32_bf16<<<(n1 + 255) / 256, 256, 0, stream>>>(w_in, winT, DIN_, H_);
    int n2 = H_ * H_;    // recurrent weights [1024,1024] -> [1024,1024]
    transpose_f32_bf16<<<(n2 + 255) / 256, 256, 0, stream>>>(w_hr, whrT, H_, H_);
    transpose_f32_bf16<<<(n2 + 255) / 256, 256, 0, stream>>>(w_hz, whzT, H_, H_);
    transpose_f32_bf16<<<(n2 + 255) / 256, 256, 0, stream>>>(w_hn, whnT, H_, H_);
    int n3 = H_ * O_;    // output weights [1024,512] -> [512,1024]
    transpose_f32_bf16<<<(n3 + 255) / 256, 256, 0, stream>>>(w_out, woT, H_, O_);
  }
  // 3. zero h, h_bf16, barrier
  {
    int n = B_ * H_;
    init_state<<<(n + 255) / 256, 256, 0, stream>>>(hf, hbm, bar, n);
  }
  // 4. input-side gate GEMMs: [BT,DIN] x [DIN,H] -> bf16 gates (bias fused)
  {
    long tiles = (long)(BT >> 4) * (H_ >> 6);
    int blocks = (int)((tiles + 7) / 8);
    gemm_bf16_nt<1><<<blocks, 256, 0, stream>>>(xbf, wirT, b_ir, gxr, (int)BT, H_, DIN_);
    gemm_bf16_nt<1><<<blocks, 256, 0, stream>>>(xbf, wizT, b_iz, gxz, (int)BT, H_, DIN_);
    gemm_bf16_nt<1><<<blocks, 256, 0, stream>>>(xbf, winT, b_in, gxn, (int)BT, H_, DIN_);
  }
  // 5. persistent recurrent scan (WMMA + TDM-staged h + grid barrier per step)
  gru_scan<<<NBLK_SCAN, 256, (size_t)B_ * H_ * 2, stream>>>(
      gxr, gxz, gxn, whrT, whzT, whnT, b_hn, hf, hbm, hsb, tmp, bar);
  // 6. output projection: [BT,H] x [H,O] -> f32 d_out (bias fused)
  {
    long tiles = (long)(BT >> 4) * (O_ >> 6);
    int blocks = (int)((tiles + 7) / 8);
    gemm_bf16_nt<0><<<blocks, 256, 0, stream>>>(hsb, woT, b_out, out, (int)BT, O_, H_);
  }
}
